// PointNetSetAbstractionMsg_26379689132422
// MI455X (gfx1250) — compile-verified, matrix operand order checked by reading, not run on hardware
//
#include <hip/hip_runtime.h>

typedef _Float16 f16;
typedef _Float16 v16h __attribute__((ext_vector_type(16)));
typedef _Float16 v8h  __attribute__((ext_vector_type(8)));
typedef float    v8f  __attribute__((ext_vector_type(8)));

static constexpr int BB = 8, NN = 4096, SS = 1024, DD = 64;
static constexpr float EPSV = 1e-5f;

union HF16 { v16h v; v8h h[2]; };

// ---------------- kernel 1: gather centers ----------------
__global__ void k_gather(const float* __restrict__ xyz, const float* __restrict__ pts,
                         const int* __restrict__ fps, float* __restrict__ out_newxyz,
                         float* __restrict__ ws_nx, float* __restrict__ ws_np) {
  int t = blockIdx.x * blockDim.x + threadIdx.x;
  if (t >= BB * SS) return;
  int b = t / SS, s = t - b * SS;
  int idx = fps[t];
  idx = idx < 0 ? 0 : (idx >= NN ? NN - 1 : idx);
  float x = xyz[(b * 3 + 0) * NN + idx];
  float y = xyz[(b * 3 + 1) * NN + idx];
  float z = xyz[(b * 3 + 2) * NN + idx];
  out_newxyz[(b * 3 + 0) * SS + s] = x;
  out_newxyz[(b * 3 + 1) * SS + s] = y;
  out_newxyz[(b * 3 + 2) * SS + s] = z;
  ws_nx[t * 3 + 0] = x; ws_nx[t * 3 + 1] = y; ws_nx[t * 3 + 2] = z;
  for (int c = 0; c < DD; ++c)
    ws_np[(size_t)t * DD + c] = pts[((size_t)b * DD + c) * NN + idx];
}

// ---------------- kernel 2: fold BN into weights (f32 -> f16, pad Cin) ----------------
__global__ void k_fold(const float* __restrict__ W, const float* __restrict__ bia,
                       const float* __restrict__ gam, const float* __restrict__ bet,
                       const float* __restrict__ mea, const float* __restrict__ var,
                       int cout, int cin, int kp,
                       f16* __restrict__ wout, float* __restrict__ bout) {
  int tot = cout * kp;
  for (int e = blockIdx.x * blockDim.x + threadIdx.x; e < tot; e += gridDim.x * blockDim.x) {
    int o = e / kp, c = e - o * kp;
    float sc = gam[o] * rsqrtf(var[o] + EPSV);
    float w = (c < cin) ? W[o * cin + c] * sc : 0.0f;
    wout[e] = (f16)w;
  }
  for (int o = blockIdx.x * blockDim.x + threadIdx.x; o < cout; o += gridDim.x * blockDim.x) {
    float sc = gam[o] * rsqrtf(var[o] + EPSV);
    bout[o] = (bia[o] - mea[o]) * sc + bet[o];
  }
}

// ---------------- kernel 3: ball query (wave32 ballot scan, first-K semantics) ----------------
template<int K>
__global__ __launch_bounds__(256) void k_query(const float* __restrict__ xyz,
                                               const float* __restrict__ ws_nx,
                                               int* __restrict__ gidx, float r2) {
  __shared__ float sx[NN], sy[NN], sz[NN];
  const int b = blockIdx.y;
  for (int i = threadIdx.x; i < NN; i += blockDim.x) {
    sx[i] = xyz[(b * 3 + 0) * NN + i];
    sy[i] = xyz[(b * 3 + 1) * NN + i];
    sz[i] = xyz[(b * 3 + 2) * NN + i];
  }
  __syncthreads();
  const int wave = threadIdx.x >> 5, lane = threadIdx.x & 31;
  for (int si = 0; si < 16; ++si) {            // 8 waves * 16 = 128 centers per block
    int s = blockIdx.x * 128 + si * 8 + wave;
    float cx = ws_nx[(b * SS + s) * 3 + 0];
    float cy = ws_nx[(b * SS + s) * 3 + 1];
    float cz = ws_nx[(b * SS + s) * 3 + 2];
    int* gout = gidx + (size_t)(b * SS + s) * K;
    int cnt = 0, firstIdx = NN - 1;
    for (int n0 = 0; n0 < NN && cnt < K; n0 += 32) {
      int n = n0 + lane;
      float dx = sx[n] - cx, dy = sy[n] - cy, dz = sz[n] - cz;
      bool in = (dx * dx + dy * dy + dz * dz) <= r2;
      unsigned m = (unsigned)__ballot(in);
      if (cnt == 0 && m) firstIdx = n0 + __ffs(m) - 1;
      int pos = cnt + __popc(m & ((1u << lane) - 1u));
      if (in && pos < K) gout[pos] = n;
      cnt += __popc(m);
    }
    int filled = cnt < K ? cnt : K;
    for (int p = filled + lane; p < K; p += 32) gout[p] = firstIdx;
  }
}

// ---------------- WMMA MLP layer (GEMM + bias + relu -> LDS f16) ----------------
template<int KP, int COUT>
__device__ __forceinline__ void mlp_layer(const f16* __restrict__ X, const f16* __restrict__ Wg,
                                          const float* __restrict__ bias, f16* __restrict__ Y,
                                          int mt, int nh, int lane) {
  constexpr int NT = COUT / 32;            // 16-col tiles per wave (wave owns half of COUT)
  const int hi = lane >> 4, ln = lane & 15;
  const int row = mt * 16 + ln;
  v8f acc[NT] = {};
  for (int kc = 0; kc < KP / 32; ++kc) {
    const int kb = kc * 32 + hi * 8;
    HF16 ua;
    ua.h[0] = *(const v8h*)(X + row * KP + kb);
    ua.h[1] = *(const v8h*)(X + row * KP + kb + 16);
    for (int t = 0; t < NT; ++t) {
      int ncol = nh * (COUT / 2) + t * 16 + ln;
      const f16* wp = Wg + (size_t)ncol * KP + kb;
      HF16 ub;
      ub.h[0] = *(const v8h*)wp;
      ub.h[1] = *(const v8h*)(wp + 16);
      acc[t] = __builtin_amdgcn_wmma_f32_16x16x32_f16(false, ua.v, false, ub.v,
                                                      (short)0, acc[t], false, false);
    }
  }
  for (int t = 0; t < NT; ++t) {
    int ncol = nh * (COUT / 2) + t * 16 + ln;
    float bs = bias[ncol];
    for (int r = 0; r < 8; ++r) {
      int orow = mt * 16 + r + hi * 8;
      float v = acc[t][r] + bs;
      v = v > 0.0f ? v : 0.0f;
      Y[orow * COUT + ncol] = (f16)v;
    }
  }
}

// ---------------- final WMMA layer fused with max-pool over K ----------------
template<int KP, int COUT, int K>
__device__ __forceinline__ void mlp_layer_max(const f16* __restrict__ X, const f16* __restrict__ Wg,
                                              const float* __restrict__ bias, unsigned* __restrict__ smax,
                                              int mt, int nh, int lane) {
  constexpr int NT = COUT / 32;
  const int hi = lane >> 4, ln = lane & 15;
  const int row = mt * 16 + ln;
  v8f acc[NT] = {};
  for (int kc = 0; kc < KP / 32; ++kc) {
    const int kb = kc * 32 + hi * 8;
    HF16 ua;
    ua.h[0] = *(const v8h*)(X + row * KP + kb);
    ua.h[1] = *(const v8h*)(X + row * KP + kb + 16);
    for (int t = 0; t < NT; ++t) {
      int ncol = nh * (COUT / 2) + t * 16 + ln;
      const f16* wp = Wg + (size_t)ncol * KP + kb;
      HF16 ub;
      ub.h[0] = *(const v8h*)wp;
      ub.h[1] = *(const v8h*)(wp + 16);
      acc[t] = __builtin_amdgcn_wmma_f32_16x16x32_f16(false, ua.v, false, ub.v,
                                                      (short)0, acc[t], false, false);
    }
  }
  const int sl = (mt * 16) / K;            // which center this M-tile belongs to
  for (int t = 0; t < NT; ++t) {
    int ncol = nh * (COUT / 2) + t * 16 + ln;
    float bs = bias[ncol];
    float m = 0.0f;                         // relu floor
    for (int r = 0; r < 8; ++r) {
      float v = acc[t][r] + bs;
      if (v > m) m = v;
    }
    float o = __shfl_xor(m, 16);           // combine rows {r} with rows {r+8}
    m = m > o ? m : o;
    if (hi == 0) atomicMax(&smax[sl * COUT + ncol], __float_as_uint(m));
  }
}

// ---------------- kernel 4: feature build + 3-layer WMMA MLP + max-pool ----------------
template<int K, int C1, int C2, int C3, int TS, int CHOFF>
__global__ __launch_bounds__(256) void k_mlp(
    const float* __restrict__ xyz, const float* __restrict__ pts,
    const float* __restrict__ ws_nx, const float* __restrict__ ws_np,
    const int* __restrict__ gidx,
    const f16* __restrict__ W1, const f16* __restrict__ W2, const f16* __restrict__ W3,
    const float* __restrict__ b1, const float* __restrict__ b2, const float* __restrict__ b3,
    float* __restrict__ out, float inv_radius) {
  constexpr int KP0 = 160;                 // 131 channels padded to 5*32
  constexpr int M = TS * K;                // 64 rows per block
  __shared__ f16 Xb[M * KP0];              // layer1 input; reused (stride C2) as layer3 input
  __shared__ f16 Yb[M * C1];               // layer1 output / layer2 input
  __shared__ unsigned smax[TS * C3];

  const int b  = blockIdx.x / (SS / TS);
  const int s0 = (blockIdx.x - b * (SS / TS)) * TS;
  const int tid = threadIdx.x;
  const int wave = tid >> 5, lane = tid & 31;
  const int mt = wave & 3, nh = wave >> 2;

  // build input features: [gpts-ctr (64) | ctr (64) | (gxyz-center)/r (3) | pad]
  for (int e = tid; e < M * KP0; e += 256) {
    int row = e / KP0, c = e - row * KP0;
    int sl = row / K, k = row - sl * K;
    int bs = b * SS + (s0 + sl);
    int g = gidx[(size_t)bs * K + k];
    float v;
    if (c < 64)        v = pts[((size_t)b * DD + c) * NN + g] - ws_np[(size_t)bs * DD + c];
    else if (c < 128)  v = ws_np[(size_t)bs * DD + (c - 64)];
    else if (c < 131)  v = (xyz[((size_t)b * 3 + (c - 128)) * NN + g] - ws_nx[bs * 3 + (c - 128)]) * inv_radius;
    else               v = 0.0f;
    Xb[e] = (f16)v;
  }
  for (int e = tid; e < TS * C3; e += 256) smax[e] = 0u;
  __syncthreads();

  mlp_layer<KP0, C1>(Xb, W1, b1, Yb, mt, nh, lane);
  __syncthreads();
  mlp_layer<C1, C2>(Yb, W2, b2, Xb, mt, nh, lane);   // output stride C2 into Xb
  __syncthreads();
  mlp_layer_max<C2, C3, K>(Xb, W3, b3, smax, mt, nh, lane);
  __syncthreads();

  for (int e = tid; e < TS * C3; e += 256) {
    int sl = e / C3, c = e - sl * C3;
    out[(size_t)b * 384 * SS + (size_t)(CHOFF + c) * SS + (s0 + sl)] = __uint_as_float(smax[e]);
  }
}

// ---------------- host side ----------------
extern "C" void kernel_launch(void* const* d_in, const int* in_sizes, int n_in,
                              void* d_out, int out_size, void* d_ws, size_t ws_size,
                              hipStream_t stream) {
  (void)in_sizes; (void)n_in; (void)out_size; (void)ws_size;
  const float* xyz = (const float*)d_in[0];
  const float* pts = (const float*)d_in[1];
  const int*   fps = (const int*)d_in[2];
  auto P = [&](int br, int ly, int j) { return (const float*)d_in[3 + (br * 3 + ly) * 6 + j]; };

  // workspace partition
  char* w = (char*)d_ws;
  auto take = [&](size_t bytes) { char* r = w; w += (bytes + 255) & ~(size_t)255; return r; };
  float* ws_nx  = (float*)take((size_t)BB * SS * 3 * sizeof(float));
  float* ws_np  = (float*)take((size_t)BB * SS * DD * sizeof(float));
  int*   gidx0  = (int*)take((size_t)BB * SS * 32 * sizeof(int));
  int*   gidx1  = (int*)take((size_t)BB * SS * 64 * sizeof(int));
  f16* w01 = (f16*)take((size_t)64  * 160 * 2);
  f16* w02 = (f16*)take((size_t)64  * 64  * 2);
  f16* w03 = (f16*)take((size_t)128 * 64  * 2);
  f16* w11 = (f16*)take((size_t)128 * 160 * 2);
  f16* w12 = (f16*)take((size_t)128 * 128 * 2);
  f16* w13 = (f16*)take((size_t)256 * 128 * 2);
  float* bb01 = (float*)take(64 * 4);
  float* bb02 = (float*)take(64 * 4);
  float* bb03 = (float*)take(128 * 4);
  float* bb11 = (float*)take(128 * 4);
  float* bb12 = (float*)take(128 * 4);
  float* bb13 = (float*)take(256 * 4);

  float* out_newxyz = (float*)d_out;
  float* out_pts    = (float*)d_out + (size_t)BB * 3 * SS;

  k_gather<<<(BB * SS + 255) / 256, 256, 0, stream>>>(xyz, pts, fps, out_newxyz, ws_nx, ws_np);

  auto fold = [&](int br, int ly, int cout, int cin, int kp, f16* wo, float* bo) {
    int grid = (cout * kp + 255) / 256;
    k_fold<<<grid, 256, 0, stream>>>(P(br, ly, 0), P(br, ly, 1), P(br, ly, 2),
                                     P(br, ly, 3), P(br, ly, 4), P(br, ly, 5),
                                     cout, cin, kp, wo, bo);
  };
  fold(0, 0, 64, 131, 160, w01, bb01);
  fold(0, 1, 64, 64, 64, w02, bb02);
  fold(0, 2, 128, 64, 64, w03, bb03);
  fold(1, 0, 128, 131, 160, w11, bb11);
  fold(1, 1, 128, 128, 128, w12, bb12);
  fold(1, 2, 256, 128, 128, w13, bb13);

  dim3 qgrid(SS / 128, BB);
  k_query<32><<<qgrid, 256, 0, stream>>>(xyz, ws_nx, gidx0, 0.2f * 0.2f);
  k_query<64><<<qgrid, 256, 0, stream>>>(xyz, ws_nx, gidx1, 0.4f * 0.4f);

  // branch 0: K=32, MLP 131->64->64->128, 2 centers per block (M = 64 rows)
  k_mlp<32, 64, 64, 128, 2, 0><<<BB * SS / 2, 256, 0, stream>>>(
      xyz, pts, ws_nx, ws_np, gidx0, w01, w02, w03, bb01, bb02, bb03, out_pts, 1.0f / 0.2f);
  // branch 1: K=64, MLP 131->128->128->256, 1 center per block (M = 64 rows)
  k_mlp<64, 128, 128, 256, 1, 128><<<BB * SS, 256, 0, stream>>>(
      xyz, pts, ws_nx, ws_np, gidx1, w11, w12, w13, bb11, bb12, bb13, out_pts, 1.0f / 0.4f);
}
